// GINModelIntegrated_73521250173226
// MI455X (gfx1250) — matrix operand order repaired and checked by reading, not compiled, not run on hardware
//
#include <hip/hip_runtime.h>

// ---------------------------------------------------------------------------
// Types for CDNA5 WMMA
// ---------------------------------------------------------------------------
typedef __attribute__((ext_vector_type(16))) __bf16 bf16x16;
typedef __attribute__((ext_vector_type(8)))  float  f32x8;

__device__ __forceinline__ unsigned short f2bf(float f) {
  // round-to-nearest-even fp32 -> bf16
  unsigned int u = __float_as_uint(f);
  return (unsigned short)((u + 0x7FFFu + ((u >> 16) & 1u)) >> 16);
}

__device__ __forceinline__ unsigned int pack2bf(float lo, float hi) {
#if __has_builtin(__builtin_amdgcn_cvt_pk_bf16_f32)
  // single v_cvt_pk_bf16_f32 (RNE) instead of ~7 VALU ops
  typedef __attribute__((ext_vector_type(2))) __bf16 bf16x2;
  union { bf16x2 v; unsigned int u; } cv;
  cv.v = __builtin_amdgcn_cvt_pk_bf16_f32(lo, hi);
  return cv.u;
#else
  return (unsigned int)f2bf(lo) | ((unsigned int)f2bf(hi) << 16);
#endif
}

// ---------------------------------------------------------------------------
// 0) Weight prep: Wt[col*K + k] = bf16(W[k*128 + col])   (transpose + convert)
//    Makes every B fragment 32 contiguous bytes -> 2x global_load_b128 (L2-hot)
// ---------------------------------------------------------------------------
__global__ void w2bf_kernel(const float* __restrict__ W,
                            unsigned short* __restrict__ Wt, int K) {
  int id = blockIdx.x * blockDim.x + threadIdx.x;   // coalesced read of W
  if (id >= K * 128) return;
  int k = id >> 7, col = id & 127;
  Wt[(size_t)col * K + k] = f2bf(W[id]);
}

// ---------------------------------------------------------------------------
// 1) Preprocess fused with agg init:
//    out = (x==0) ? me : x ;  agg = (2+eps)*out     (float4 granularity)
// ---------------------------------------------------------------------------
template <int F>
__global__ void preprocess_kernel(const float* __restrict__ in,
                                  const float* __restrict__ me,
                                  float* __restrict__ out,
                                  float* __restrict__ agg,
                                  const float* __restrict__ eps, long n4) {
  long i4 = (long)blockIdx.x * blockDim.x + threadIdx.x;
  if (i4 >= n4) return;
  float s = 2.0f + *eps;
  float4 v = ((const float4*)in)[i4];
  int col = (int)((i4 * 4) % F);
  v.x = (v.x == 0.0f) ? me[col + 0] : v.x;
  v.y = (v.y == 0.0f) ? me[col + 1] : v.y;
  v.z = (v.z == 0.0f) ? me[col + 2] : v.z;
  v.w = (v.w == 0.0f) ? me[col + 3] : v.w;
  ((float4*)out)[i4] = v;
  float4 a = make_float4(s * v.x, s * v.y, s * v.z, s * v.w);
  ((float4*)agg)[i4] = a;
}

// ---------------------------------------------------------------------------
// 2) scatter-add over real edges: agg[dst] += h[src]  (float4 chunks)
// ---------------------------------------------------------------------------
template <int F>
__global__ void scatter_add_kernel(const float* __restrict__ h,
                                   const int* __restrict__ src,
                                   const int* __restrict__ dst,
                                   float* __restrict__ agg, int E) {
  constexpr int CH = F / 4;
  long tid = (long)blockIdx.x * blockDim.x + threadIdx.x;
  if (tid >= (long)E * CH) return;
  int e  = (int)(tid / CH);
  int c4 = (int)(tid % CH) * 4;
  int s = src[e], d = dst[e];
  float4 v = *(const float4*)&h[(size_t)s * F + c4];
  float* p = &agg[(size_t)d * F + c4];
  unsafeAtomicAdd(p + 0, v.x);
  unsafeAtomicAdd(p + 1, v.y);
  unsafeAtomicAdd(p + 2, v.z);
  unsafeAtomicAdd(p + 3, v.w);
}

// ---------------------------------------------------------------------------
// 3) WMMA GEMM (LDS-free): out[N,128] = lrelu(A[N,K] @ W[K,128] + bias)
//    Optionally fuses next layer's agg init: agg = (2+eps)*out.
//    Each wave owns a 16-row stripe (A rows are wave-private -> no LDS);
//    B fragments come straight from pre-transposed bf16 Wt (L2-resident).
//    OOB rows: A pointer clamped (one cndmask, no per-step exec divergence);
//    stores remain range-guarded.
// ---------------------------------------------------------------------------
template <int K>
__global__ __launch_bounds__(256)
void gin_gemm_kernel(const float* __restrict__ A,
                     const unsigned short* __restrict__ Wt,
                     const float* __restrict__ bias, float* __restrict__ out,
                     int Nrows, float slope,
                     float* __restrict__ agg, const float* __restrict__ eps) {
  const int lane = threadIdx.x & 31;
  const int wave = threadIdx.x >> 5;
  const int m16  = lane & 15;   // A: M row   | B/D: N col
  const int hh   = lane >> 4;   // K sub-range selector
  const int tilerow = (blockIdx.x * 8 + wave) * 16;
  const int arow = tilerow + m16;
  const int arowc = (arow < Nrows) ? arow : (Nrows - 1);  // clamp, not branch
  const float* ap = A + (size_t)arowc * K;

  f32x8 acc[8];
#pragma unroll
  for (int i = 0; i < 8; ++i)
#pragma unroll
    for (int r = 0; r < 8; ++r) acc[i][r] = 0.0f;

#pragma unroll
  for (int k0 = 0; k0 < K; k0 += 32) {
    // A fragment (contiguous K order, matches B):
    //   e0..15 -> K = k0 + hh*16 + e
    float4 a0 = *(const float4*)(ap + k0 + hh * 16);
    float4 a1 = *(const float4*)(ap + k0 + hh * 16 + 4);
    float4 a2 = *(const float4*)(ap + k0 + hh * 16 + 8);
    float4 a3 = *(const float4*)(ap + k0 + hh * 16 + 12);

    union { unsigned int u[8]; bf16x16 v; } af;
    af.u[0] = pack2bf(a0.x, a0.y); af.u[1] = pack2bf(a0.z, a0.w);
    af.u[2] = pack2bf(a1.x, a1.y); af.u[3] = pack2bf(a1.z, a1.w);
    af.u[4] = pack2bf(a2.x, a2.y); af.u[5] = pack2bf(a2.z, a2.w);
    af.u[6] = pack2bf(a3.x, a3.y); af.u[7] = pack2bf(a3.z, a3.w);

#pragma unroll
    for (int ct = 0; ct < 8; ++ct) {
      // B fragment: N = lane%16, K = k0 + (lane/16)*16 + e  -> contiguous in Wt
      const unsigned short* bp = Wt + (size_t)(ct * 16 + m16) * K + k0 + hh * 16;
      union { uint4 q[2]; bf16x16 v; } bf;
      bf.q[0] = *(const uint4*)bp;
      bf.q[1] = *(const uint4*)(bp + 8);
      acc[ct] = __builtin_amdgcn_wmma_f32_16x16x32_bf16(
          false, af.v, false, bf.v, (short)0, acc[ct], false, false);
    }
  }

  // epilogue: D layout -> N = lane%16, M = r + 8*(lane/16)
  float s = 0.0f;
  if (agg) s = 2.0f + *eps;
#pragma unroll
  for (int ct = 0; ct < 8; ++ct) {
    int col = ct * 16 + m16;
    float bz = bias[col];
#pragma unroll
    for (int r = 0; r < 8; ++r) {
      int grow = tilerow + r + 8 * hh;
      if (grow < Nrows) {
        float v = acc[ct][r] + bz;
        float o = (v > 0.0f) ? v : slope * v;
        out[(size_t)grow * 128 + col] = o;
        if (agg) agg[(size_t)grow * 128 + col] = s * o;
      }
    }
  }
}

// ---------------------------------------------------------------------------
// 4) mixer: out[N,8] = lrelu(concat(xe,xc)[N,256] @ Wm[256,8] + bm)
// ---------------------------------------------------------------------------
__global__ __launch_bounds__(256)
void mixer_kernel(const float* __restrict__ xe, const float* __restrict__ xc,
                  const float* __restrict__ Wm, const float* __restrict__ bm,
                  float* __restrict__ out, int n) {
  __shared__ float wl[256 * 8];
  for (int i = threadIdx.x; i < 2048; i += 256) wl[i] = Wm[i];
  __syncthreads();

  const int lane = threadIdx.x & 31;
  const int wave = threadIdx.x >> 5;
  const int node = blockIdx.x * 8 + wave;
  if (node >= n) return;

  float4 e4 = *(const float4*)&xe[(size_t)node * 128 + lane * 4];
  float4 c4 = *(const float4*)&xc[(size_t)node * 128 + lane * 4];
  float ev[4] = {e4.x, e4.y, e4.z, e4.w};
  float cv[4] = {c4.x, c4.y, c4.z, c4.w};

  float part[8];
#pragma unroll
  for (int m = 0; m < 8; ++m) part[m] = 0.0f;
#pragma unroll
  for (int j = 0; j < 4; ++j) {
    int fe = lane * 4 + j;
#pragma unroll
    for (int m = 0; m < 8; ++m) {
      part[m] += ev[j] * wl[fe * 8 + m];
      part[m] += cv[j] * wl[(128 + fe) * 8 + m];
    }
  }
#pragma unroll
  for (int m = 0; m < 8; ++m)
#pragma unroll
    for (int off = 16; off >= 1; off >>= 1)
      part[m] += __shfl_xor(part[m], off, 32);

  if (lane == 0) {
#pragma unroll
    for (int m = 0; m < 8; ++m) {
      float v = part[m] + bm[m];
      out[(size_t)node * 8 + m] = (v > 0.0f) ? v : 0.2f * v;
    }
  }
}

// ---------------------------------------------------------------------------
// 5) head: logits = relu(out[cidx] @ Wp1 + bp1) @ Wp2 + bp2   (wave per node)
// ---------------------------------------------------------------------------
__global__ __launch_bounds__(256)
void head_kernel(const float* __restrict__ feat, const int* __restrict__ cidx,
                 const float* __restrict__ Wp1, const float* __restrict__ bp1,
                 const float* __restrict__ Wp2, const float* __restrict__ bp2,
                 float* __restrict__ logits, int C) {
  const int lane = threadIdx.x & 31;
  const int wave = threadIdx.x >> 5;
  const int n = blockIdx.x * 8 + wave;
  if (n >= C) return;
  const int node = cidx[n];

  float c8[8];
#pragma unroll
  for (int k = 0; k < 8; ++k) c8[k] = feat[(size_t)node * 8 + k];

  float part[8];
#pragma unroll
  for (int m = 0; m < 8; ++m) part[m] = 0.0f;

#pragma unroll
  for (int jj = 0; jj < 4; ++jj) {
    int j = lane + jj * 32;
    float t = bp1[j];
#pragma unroll
    for (int k = 0; k < 8; ++k) t += c8[k] * Wp1[k * 128 + j];
    t = fmaxf(t, 0.0f);
#pragma unroll
    for (int m = 0; m < 8; ++m) part[m] += t * Wp2[j * 8 + m];
  }
#pragma unroll
  for (int m = 0; m < 8; ++m)
#pragma unroll
    for (int off = 16; off >= 1; off >>= 1)
      part[m] += __shfl_xor(part[m], off, 32);

  if (lane == 0) {
#pragma unroll
    for (int m = 0; m < 8; ++m) logits[(size_t)n * 8 + m] = part[m] + bp2[m];
  }
}

// ---------------------------------------------------------------------------
// launch
// ---------------------------------------------------------------------------
extern "C" void kernel_launch(void* const* d_in, const int* in_sizes, int n_in,
                              void* d_out, int out_size, void* d_ws, size_t ws_size,
                              hipStream_t stream) {
  const float* x    = (const float*)d_in[0];
  const float* c    = (const float*)d_in[1];
  const int*   eidx = (const int*)d_in[2];
  const int*   cidx = (const int*)d_in[3];
  const float* me_x = (const float*)d_in[4];
  const float* me_c = (const float*)d_in[5];
  const float* W1e = (const float*)d_in[6];  const float* b1e = (const float*)d_in[7];  const float* eps1e = (const float*)d_in[8];
  const float* W2e = (const float*)d_in[9];  const float* b2e = (const float*)d_in[10]; const float* eps2e = (const float*)d_in[11];
  const float* W1c = (const float*)d_in[12]; const float* b1c = (const float*)d_in[13]; const float* eps1c = (const float*)d_in[14];
  const float* W2c = (const float*)d_in[15]; const float* b2c = (const float*)d_in[16]; const float* eps2c = (const float*)d_in[17];
  const float* Wm  = (const float*)d_in[18]; const float* bm  = (const float*)d_in[19];
  const float* Wp1 = (const float*)d_in[20]; const float* bp1 = (const float*)d_in[21];
  const float* Wp2 = (const float*)d_in[22]; const float* bp2 = (const float*)d_in[23];
  float* out = (float*)d_out;

  const int N = in_sizes[0] / 256;
  const int E = in_sizes[2] / 2;
  const int C = in_sizes[3];
  const int* src = eidx;
  const int* dst = eidx + E;

  // ws layout: bf16 weights first (16B-aligned region), then fp32 buffers
  unsigned short* Wt1e = (unsigned short*)d_ws;       // 256*128
  unsigned short* Wt2e = Wt1e + 256 * 128;            // 128*128
  unsigned short* Wt1c = Wt2e + 128 * 128;            // 128*128
  unsigned short* Wt2c = Wt1c + 128 * 128;            // 128*128
  float* fbase = (float*)(Wt2c + 128 * 128);
  float* bufA = fbase;                       // [N,256] preprocessed features
  float* bufB = bufA + (size_t)N * 256;      // [N,256] agg ping
  float* bufC = bufB + (size_t)N * 256;      // [N,128] hidden layer 1
  float* bufD = bufC + (size_t)N * 128;      // [N,128] x_exp (persists)
  float* bufE = bufD + (size_t)N * 128;      // [N,128] x_cnv (persists)
  float* bufF = bufE + (size_t)N * 128;      // [N,128] agg pong

  auto cdiv = [](long a, long b) { return (int)((a + b - 1) / b); };
  const int TB = 256;
  const int gemmBlocks = cdiv(N, 128);

  // ---- bf16 transposed weight copies (once per launch, tiny) ----
  w2bf_kernel<<<cdiv(256 * 128, TB), TB, 0, stream>>>(W1e, Wt1e, 256);
  w2bf_kernel<<<cdiv(128 * 128, TB), TB, 0, stream>>>(W2e, Wt2e, 128);
  w2bf_kernel<<<cdiv(128 * 128, TB), TB, 0, stream>>>(W1c, Wt1c, 128);
  w2bf_kernel<<<cdiv(128 * 128, TB), TB, 0, stream>>>(W2c, Wt2c, 128);

  // ---- expression stream (FE=256) ----
  preprocess_kernel<256><<<cdiv((long)N * 64, TB), TB, 0, stream>>>(x, me_x, bufA, bufB, eps1e, (long)N * 64);
  scatter_add_kernel<256><<<cdiv((long)E * 64, TB), TB, 0, stream>>>(bufA, src, dst, bufB, E);
  gin_gemm_kernel<256><<<gemmBlocks, TB, 0, stream>>>(bufB, Wt1e, b1e, bufC, N, 0.2f, bufF, eps2e);
  scatter_add_kernel<128><<<cdiv((long)E * 32, TB), TB, 0, stream>>>(bufC, src, dst, bufF, E);
  gin_gemm_kernel<128><<<gemmBlocks, TB, 0, stream>>>(bufF, Wt2e, b2e, bufD, N, 0.2f, nullptr, nullptr);

  // ---- CNV stream (FC=128) ----
  preprocess_kernel<128><<<cdiv((long)N * 32, TB), TB, 0, stream>>>(c, me_c, bufA, bufB, eps1c, (long)N * 32);
  scatter_add_kernel<128><<<cdiv((long)E * 32, TB), TB, 0, stream>>>(bufA, src, dst, bufB, E);
  gin_gemm_kernel<128><<<gemmBlocks, TB, 0, stream>>>(bufB, Wt1c, b1c, bufC, N, 0.2f, bufF, eps2c);
  scatter_add_kernel<128><<<cdiv((long)E * 32, TB), TB, 0, stream>>>(bufC, src, dst, bufF, E);
  gin_gemm_kernel<128><<<gemmBlocks, TB, 0, stream>>>(bufF, Wt2c, b2c, bufE, N, 0.2f, nullptr, nullptr);

  // ---- mixer + head ----
  mixer_kernel<<<cdiv(N, 8), TB, 0, stream>>>(bufD, bufE, Wm, bm, out, N);
  head_kernel<<<cdiv(C, 8), TB, 0, stream>>>(out, cidx, Wp1, bp1, Wp2, bp2,
                                             out + (size_t)N * 8, C);
}
